// Decoder_new_84602265797073
// MI455X (gfx1250) — matrix operand order repaired and checked by reading, not compiled, not run on hardware
//
#include <hip/hip_runtime.h>
#include <hip/hip_bf16.h>

typedef __bf16 bf16;
typedef __attribute__((ext_vector_type(16))) __bf16 v16bf;
typedef __attribute__((ext_vector_type(8)))  __bf16 v8bf;
typedef __attribute__((ext_vector_type(8)))  float  v8f;

static constexpr int BB = 16;      // batch
static constexpr int NN = 1024;    // nodes
static constexpr int HH = 64;      // hidden
static constexpr int EE = 32768;   // edges
static constexpr int LL = 4;       // layers
static constexpr int K1 = 160;     // padded 2H+1 (129 -> 160, 5 k-steps of 32)

__device__ __forceinline__ float silu_f(float x) { return x / (1.0f + __expf(-x)); }

__device__ __forceinline__ v16bf frag_cat(v8bf lo, v8bf hi) {
    return __builtin_shufflevector(lo, hi, 0,1,2,3,4,5,6,7,8,9,10,11,12,13,14,15);
}

// A-fragment: rowbase points at A[m][kbase]; lane m=(l&15), kh=(l>>4).
// elems 0..7  <- K = kbase + kh*16 + i
// elems 8..15 <- K = kbase + kh*16 + 8 + i
__device__ __forceinline__ v16bf load_afrag(const bf16* rowbase, int kh) {
    const bf16* p = rowbase + kh * 16;
    v8bf lo = *(const v8bf*)(p);
    v8bf hi = *(const v8bf*)(p + 8);
    return frag_cat(lo, hi);
}

// Same A-fragment shape but source is fp32 (converted to bf16 in regs).
__device__ __forceinline__ v16bf load_afrag_f32(const float* rowbase, int kh) {
    const float* p = rowbase + kh * 16;
    v16bf r;
#pragma unroll
    for (int i = 0; i < 8; ++i) r[i] = (bf16)p[i];
#pragma unroll
    for (int i = 0; i < 8; ++i) r[8 + i] = (bf16)p[8 + i];
    return r;
}

// B-fragment: wb points at W[n][kbase + kh*16] in packed [n][ldk] layout.
// elem i <- K = kbase + kh*16 + i (16 contiguous bf16)
__device__ __forceinline__ v16bf load_bfrag(const bf16* wb) {
    v8bf lo = *(const v8bf*)(wb);
    v8bf hi = *(const v8bf*)(wb + 8);
    return frag_cat(lo, hi);
}

#define WMMA_BF16(A, Bm, C) \
    __builtin_amdgcn_wmma_f32_16x16x32_bf16(false, (A), false, (Bm), (short)0, (C), false, false)

__device__ __forceinline__ v8f zero_v8f() {
    v8f a;
#pragma unroll
    for (int i = 0; i < 8; ++i) a[i] = 0.0f;
    return a;
}

// ---------------------------------------------------------------- prep kernels

// src: [nmat][K][64] fp32 (k-major) -> dst: [nmat][64][ldk] bf16 (k padded w/ 0)
__global__ void pack_w(const float* __restrict__ src, bf16* __restrict__ dst,
                       int K, int ldk, int nmat) {
    int idx = blockIdx.x * blockDim.x + threadIdx.x;
    int total = nmat * 64 * ldk;
    if (idx >= total) return;
    int k  = idx % ldk;
    int n  = (idx / ldk) % 64;
    int mt = idx / (ldk * 64);
    dst[idx] = (k < K) ? (bf16)src[(mt * K + k) * 64 + n] : (bf16)0.0f;
}

__global__ void g_kernel(const float* __restrict__ z, const float* __restrict__ Wg,
                         const float* __restrict__ bg, float* __restrict__ g) {
    int idx = blockIdx.x * blockDim.x + threadIdx.x;
    if (idx >= BB * HH) return;
    int b = idx / HH, o = idx % HH;
    float a = bg[o];
    for (int k = 0; k < 128; ++k) a += z[b * 128 + k] * Wg[k * HH + o];
    g[idx] = a;
}

__global__ void hinit_kernel(const float* __restrict__ nf, const float* __restrict__ Wn,
                             const float* __restrict__ bn, const float* __restrict__ g,
                             float* __restrict__ hf, bf16* __restrict__ hb) {
    int idx = blockIdx.x * blockDim.x + threadIdx.x;
    if (idx >= BB * NN * HH) return;
    int o    = idx % HH;
    int node = (idx / HH) % NN;
    int b    = idx / (HH * NN);
    float a = bn[o] + g[b * HH + o]
            + nf[node * 3 + 0] * Wn[0 * HH + o]
            + nf[node * 3 + 1] * Wn[1 * HH + o]
            + nf[node * 3 + 2] * Wn[2 * HH + o];
    hf[idx] = a;
    hb[idx] = (bf16)a;
}

__global__ void counts_kernel(const int* __restrict__ edges, float* __restrict__ counts) {
    int e = blockIdx.x * blockDim.x + threadIdx.x;
    if (e < EE) atomicAdd(&counts[edges[e]], 1.0f);   // row = edge_index[0,:]
}

__global__ void posupd_kernel(float* __restrict__ pos, const float* __restrict__ pacc,
                              const float* __restrict__ counts) {
    int idx = blockIdx.x * blockDim.x + threadIdx.x;
    if (idx >= BB * NN * 3) return;
    int node = (idx / 3) % NN;
    float c = counts[node];
    c = c < 1.0f ? 1.0f : c;
    pos[idx] += pacc[idx] / c;
}

// ---------------------------------------------------------------- edge kernel
// One wave == one edge, M rows == 16 batches. GEMM chain:
//   m1 = silu([hi|hj|d2] @ eW1 + eb1)    (K=129->160, 5 ksteps, 4 ntiles)
//   m2 = silu(m1 @ eW2 + eb2)            -> atomic scatter into agg[b, row, :]
//   w  = silu(m2 @ cW1 + cb1) @ cW2      -> trans = diff*w -> scatter pos_acc
__global__ __launch_bounds__(256) void edge_kernel(
    const bf16* __restrict__ hb, const float* __restrict__ pos,
    const int* __restrict__ edges,
    const bf16* __restrict__ eW1p, const float* __restrict__ eb1,
    const bf16* __restrict__ eW2p, const float* __restrict__ eb2,
    const bf16* __restrict__ cW1p, const float* __restrict__ cb1,
    const float* __restrict__ cW2,
    float* __restrict__ agg, float* __restrict__ pacc) {
    const int lane = threadIdx.x & 31;
    const int m    = lane & 15;     // M row (batch) for A, N col for B
    const int kh   = lane >> 4;
    const int wid  = threadIdx.x >> 5;
    const int wave = blockIdx.x * (blockDim.x >> 5) + wid;
    const int nwaves = gridDim.x * (blockDim.x >> 5);

    __shared__ bf16 sm1[8][16 * 64];
    __shared__ bf16 sm2[8][16 * 64];
    bf16* m1 = sm1[wid];
    bf16* m2 = sm2[wid];

    for (int e = wave; e < EE; e += nwaves) {
        const int er = edges[e];
        const int ec = edges[EE + e];

        // per-lane batch b = m : position diff + d2
        const float* pr = pos + ((size_t)(m * NN + er)) * 3;
        const float* pc = pos + ((size_t)(m * NN + ec)) * 3;
        float dx = pr[0] - pc[0];
        float dy = pr[1] - pc[1];
        float dz = pr[2] - pc[2];
        float d2 = dx * dx + dy * dy + dz * dz;

        const bf16* hi = hb + ((size_t)(m * NN + er)) * HH;
        const bf16* hj = hb + ((size_t)(m * NN + ec)) * HH;

        v16bf aF[5];
        aF[0] = load_afrag(hi, kh);
        aF[1] = load_afrag(hi + 32, kh);
        aF[2] = load_afrag(hj, kh);
        aF[3] = load_afrag(hj + 32, kh);
        {
            v16bf a4;
#pragma unroll
            for (int i = 0; i < 16; ++i) a4[i] = (bf16)0.0f;
            if (kh == 0) a4[0] = (bf16)d2;   // K=128 column
            aF[4] = a4;
        }

        // ---- GEMM 1: [16x160] @ [160x64]
#pragma unroll
        for (int nt = 0; nt < 4; ++nt) {
            v8f acc = zero_v8f();
            const bf16* wb = eW1p + (size_t)(nt * 16 + m) * K1 + kh * 16;
#pragma unroll
            for (int ks = 0; ks < 5; ++ks)
                acc = WMMA_BF16(aF[ks], load_bfrag(wb + ks * 32), acc);
            const float bia = eb1[nt * 16 + m];
#pragma unroll
            for (int rr = 0; rr < 8; ++rr) {
                float x = silu_f(acc[rr] + bia);
                m1[(kh * 8 + rr) * 64 + nt * 16 + m] = (bf16)x;
            }
        }

        // ---- GEMM 2: m1 @ eW2 -> m2 (scatter into agg)
        v16bf a2lo = load_afrag(m1 + m * 64, kh);
        v16bf a2hi = load_afrag(m1 + m * 64 + 32, kh);
#pragma unroll
        for (int nt = 0; nt < 4; ++nt) {
            v8f acc = zero_v8f();
            const bf16* wb = eW2p + (size_t)(nt * 16 + m) * 64 + kh * 16;
            acc = WMMA_BF16(a2lo, load_bfrag(wb), acc);
            acc = WMMA_BF16(a2hi, load_bfrag(wb + 32), acc);
            const float bia = eb2[nt * 16 + m];
#pragma unroll
            for (int rr = 0; rr < 8; ++rr) {
                float x = silu_f(acc[rr] + bia);
                const int rowb = kh * 8 + rr;           // batch index
                const int colh = nt * 16 + m;
                m2[rowb * 64 + colh] = (bf16)x;
                atomicAdd(&agg[((size_t)(rowb * NN + er)) * HH + colh], x);
            }
        }

        // ---- GEMM 3 + cW2 dot: w[b]
        v16bf a3lo = load_afrag(m2 + m * 64, kh);
        v16bf a3hi = load_afrag(m2 + m * 64 + 32, kh);
        float wacc[8];
#pragma unroll
        for (int rr = 0; rr < 8; ++rr) wacc[rr] = 0.0f;
#pragma unroll
        for (int nt = 0; nt < 4; ++nt) {
            v8f acc = zero_v8f();
            const bf16* wb = cW1p + (size_t)(nt * 16 + m) * 64 + kh * 16;
            acc = WMMA_BF16(a3lo, load_bfrag(wb), acc);
            acc = WMMA_BF16(a3hi, load_bfrag(wb + 32), acc);
            const float bia = cb1[nt * 16 + m];
            const float cw  = cW2[nt * 16 + m];
#pragma unroll
            for (int rr = 0; rr < 8; ++rr)
                wacc[rr] += silu_f(acc[rr] + bia) * cw;
        }
        // reduce over the 16 columns held by each 16-lane half-group
#pragma unroll
        for (int rr = 0; rr < 8; ++rr) {
            float v = wacc[rr];
            v += __shfl_xor(v, 1);
            v += __shfl_xor(v, 2);
            v += __shfl_xor(v, 4);
            v += __shfl_xor(v, 8);
            wacc[rr] = v;
        }
        // scatter trans = diff * w into pos_acc (lane 0 covers b=0..7, lane 16 b=8..15)
#pragma unroll
        for (int rr = 0; rr < 8; ++rr) {
            const int b = kh * 8 + rr;
            float ddx = __shfl(dx, b);
            float ddy = __shfl(dy, b);
            float ddz = __shfl(dz, b);
            if (m == 0) {
                const float w = wacc[rr];
                float* p = pacc + ((size_t)(b * NN + er)) * 3;
                atomicAdd(&p[0], ddx * w);
                atomicAdd(&p[1], ddy * w);
                atomicAdd(&p[2], ddz * w);
            }
        }
    }
}

// ---------------------------------------------------------------- node kernel
// One wave == 16 consecutive nodes of one batch.
//   u = silu([h|agg] @ nW1 + nb1) @ nW2 + nb2 ; h = LN(h + u) * lng + lnb
__global__ __launch_bounds__(256) void node_kernel(
    bf16* __restrict__ hb, const float* __restrict__ agg, float* __restrict__ hf,
    const bf16* __restrict__ nW1p, const float* __restrict__ nb1,
    const bf16* __restrict__ nW2p, const float* __restrict__ nb2,
    const float* __restrict__ lng, const float* __restrict__ lnb) {
    const int lane = threadIdx.x & 31;
    const int m    = lane & 15;
    const int kh   = lane >> 4;
    const int wid  = threadIdx.x >> 5;
    const int wave = blockIdx.x * (blockDim.x >> 5) + wid;
    const int nwaves = gridDim.x * (blockDim.x >> 5);

    __shared__ bf16 su1[8][16 * 64];
    bf16* u1 = su1[wid];

    for (int tile = wave; tile < BB * NN / 16; tile += nwaves) {
        const int b     = tile >> 6;
        const int node0 = (tile & 63) << 4;
        const size_t rowbase = (size_t)(b * NN + node0 + m) * HH;

        v16bf aF[4];
        aF[0] = load_afrag(hb + rowbase, kh);
        aF[1] = load_afrag(hb + rowbase + 32, kh);
        aF[2] = load_afrag_f32(agg + rowbase, kh);
        aF[3] = load_afrag_f32(agg + rowbase + 32, kh);

#pragma unroll
        for (int nt = 0; nt < 4; ++nt) {
            v8f acc = zero_v8f();
            const bf16* wb = nW1p + (size_t)(nt * 16 + m) * 128 + kh * 16;
#pragma unroll
            for (int ks = 0; ks < 4; ++ks)
                acc = WMMA_BF16(aF[ks], load_bfrag(wb + ks * 32), acc);
            const float bia = nb1[nt * 16 + m];
#pragma unroll
            for (int rr = 0; rr < 8; ++rr) {
                float x = silu_f(acc[rr] + bia);
                u1[(kh * 8 + rr) * 64 + nt * 16 + m] = (bf16)x;
            }
        }

        v16bf a2lo = load_afrag(u1 + m * 64, kh);
        v16bf a2hi = load_afrag(u1 + m * 64 + 32, kh);

        float hn[4][8];
#pragma unroll
        for (int nt = 0; nt < 4; ++nt) {
            v8f acc = zero_v8f();
            const bf16* wb = nW2p + (size_t)(nt * 16 + m) * 64 + kh * 16;
            acc = WMMA_BF16(a2lo, load_bfrag(wb), acc);
            acc = WMMA_BF16(a2hi, load_bfrag(wb + 32), acc);
            const float bia = nb2[nt * 16 + m];
#pragma unroll
            for (int rr = 0; rr < 8; ++rr) {
                const int row = kh * 8 + rr;
                const int col = nt * 16 + m;
                hn[nt][rr] = hf[(size_t)(b * NN + node0 + row) * HH + col] + acc[rr] + bia;
            }
        }

        // LayerNorm over 64 cols per row (4 ntiles in regs x 16 lanes via shfl)
        float mu[8], inv[8];
#pragma unroll
        for (int rr = 0; rr < 8; ++rr) {
            float s = hn[0][rr] + hn[1][rr] + hn[2][rr] + hn[3][rr];
            s += __shfl_xor(s, 1); s += __shfl_xor(s, 2);
            s += __shfl_xor(s, 4); s += __shfl_xor(s, 8);
            mu[rr] = s * (1.0f / 64.0f);
        }
#pragma unroll
        for (int rr = 0; rr < 8; ++rr) {
            float q = 0.0f;
#pragma unroll
            for (int nt = 0; nt < 4; ++nt) {
                float d = hn[nt][rr] - mu[rr];
                q += d * d;
            }
            q += __shfl_xor(q, 1); q += __shfl_xor(q, 2);
            q += __shfl_xor(q, 4); q += __shfl_xor(q, 8);
            inv[rr] = rsqrtf(q * (1.0f / 64.0f) + 1e-5f);
        }
#pragma unroll
        for (int nt = 0; nt < 4; ++nt) {
            const int col = nt * 16 + m;
            const float gs = lng[col], bs = lnb[col];
#pragma unroll
            for (int rr = 0; rr < 8; ++rr) {
                const int row = kh * 8 + rr;
                float y = (hn[nt][rr] - mu[rr]) * inv[rr] * gs + bs;
                const size_t idx = (size_t)(b * NN + node0 + row) * HH + col;
                hf[idx] = y;
                hb[idx] = (bf16)y;
            }
        }
    }
}

// ---------------------------------------------------------------- output head
// out = relu(h @ oW1 + ob1) @ oW2 + ob2
__global__ __launch_bounds__(256) void out_kernel(
    const bf16* __restrict__ hb,
    const bf16* __restrict__ oW1p, const float* __restrict__ ob1,
    const float* __restrict__ oW2, const float* __restrict__ ob2,
    float* __restrict__ out) {
    const int lane = threadIdx.x & 31;
    const int m    = lane & 15;
    const int kh   = lane >> 4;
    const int wid  = threadIdx.x >> 5;
    const int wave = blockIdx.x * (blockDim.x >> 5) + wid;
    const int nwaves = gridDim.x * (blockDim.x >> 5);

    __shared__ float sr[8][16 * 64];
    float* r1 = sr[wid];

    for (int tile = wave; tile < BB * NN / 16; tile += nwaves) {
        const int b     = tile >> 6;
        const int node0 = (tile & 63) << 4;
        const size_t rowbase = (size_t)(b * NN + node0 + m) * HH;

        v16bf alo = load_afrag(hb + rowbase, kh);
        v16bf ahi = load_afrag(hb + rowbase + 32, kh);

#pragma unroll
        for (int nt = 0; nt < 4; ++nt) {
            v8f acc = zero_v8f();
            const bf16* wb = oW1p + (size_t)(nt * 16 + m) * 64 + kh * 16;
            acc = WMMA_BF16(alo, load_bfrag(wb), acc);
            acc = WMMA_BF16(ahi, load_bfrag(wb + 32), acc);
            const float bia = ob1[nt * 16 + m];
#pragma unroll
            for (int rr = 0; rr < 8; ++rr) {
                float x = acc[rr] + bia;
                r1[(kh * 8 + rr) * 64 + nt * 16 + m] = x > 0.0f ? x : 0.0f;
            }
        }
        // small N=3 head in VALU
        if (lane < 16) {
            const int row = lane;
#pragma unroll
            for (int j = 0; j < 3; ++j) {
                float a = ob2[j];
                for (int k = 0; k < 64; ++k) a += r1[row * 64 + k] * oW2[k * 3 + j];
                out[(size_t)(b * NN + node0 + row) * 3 + j] = a;
            }
        }
    }
}

// ---------------------------------------------------------------- launch

extern "C" void kernel_launch(void* const* d_in, const int* in_sizes, int n_in,
                              void* d_out, int out_size, void* d_ws, size_t ws_size,
                              hipStream_t stream) {
    const float* z   = (const float*)d_in[0];
    const int*   ei  = (const int*)d_in[1];
    const float* nf  = (const float*)d_in[2];
    const float* Wg  = (const float*)d_in[3];
    const float* bg  = (const float*)d_in[4];
    const float* Wn  = (const float*)d_in[5];
    const float* bn  = (const float*)d_in[6];
    const float* eW1 = (const float*)d_in[7];
    const float* eb1 = (const float*)d_in[8];
    const float* eW2 = (const float*)d_in[9];
    const float* eb2 = (const float*)d_in[10];
    const float* cW1 = (const float*)d_in[11];
    const float* cb1 = (const float*)d_in[12];
    const float* cW2 = (const float*)d_in[13];
    const float* nW1 = (const float*)d_in[14];
    const float* nb1 = (const float*)d_in[15];
    const float* nW2 = (const float*)d_in[16];
    const float* nb2 = (const float*)d_in[17];
    const float* lng = (const float*)d_in[18];
    const float* lnb = (const float*)d_in[19];
    const float* oW1 = (const float*)d_in[20];
    const float* ob1 = (const float*)d_in[21];
    const float* oW2 = (const float*)d_in[22];
    const float* ob2 = (const float*)d_in[23];
    float* out = (float*)d_out;

    char* ws = (char*)d_ws;
    size_t off = 0;
    auto take = [&](size_t bytes) {
        size_t o = off;
        off = (off + bytes + 255) & ~(size_t)255;
        return o;
    };
    float* hf   = (float*)(ws + take((size_t)BB * NN * HH * 4));
    float* agg  = (float*)(ws + take((size_t)BB * NN * HH * 4));
    float* pos  = (float*)(ws + take((size_t)BB * NN * 3 * 4));
    float* pacc = (float*)(ws + take((size_t)BB * NN * 3 * 4));
    float* cnts = (float*)(ws + take((size_t)NN * 4));
    float* gbuf = (float*)(ws + take((size_t)BB * HH * 4));
    bf16* hb    = (bf16*)(ws + take((size_t)BB * NN * HH * 2));
    bf16* eW1p  = (bf16*)(ws + take((size_t)LL * 64 * K1 * 2));
    bf16* eW2p  = (bf16*)(ws + take((size_t)LL * 64 * 64 * 2));
    bf16* cW1p  = (bf16*)(ws + take((size_t)LL * 64 * 64 * 2));
    bf16* nW1p  = (bf16*)(ws + take((size_t)LL * 64 * 128 * 2));
    bf16* nW2p  = (bf16*)(ws + take((size_t)LL * 64 * 64 * 2));
    bf16* oW1p  = (bf16*)(ws + take((size_t)64 * 64 * 2));

    auto cdiv = [](int a, int b) { return (a + b - 1) / b; };

    // weight packing (bf16, column-packed [n][k] with k-padding)
    pack_w<<<cdiv(LL * 64 * K1, 256), 256, 0, stream>>>(eW1, eW1p, 129, K1, LL);
    pack_w<<<cdiv(LL * 64 * 64, 256), 256, 0, stream>>>(eW2, eW2p, 64, 64, LL);
    pack_w<<<cdiv(LL * 64 * 64, 256), 256, 0, stream>>>(cW1, cW1p, 64, 64, LL);
    pack_w<<<cdiv(LL * 64 * 128, 256), 256, 0, stream>>>(nW1, nW1p, 128, 128, LL);
    pack_w<<<cdiv(LL * 64 * 64, 256), 256, 0, stream>>>(nW2, nW2p, 64, 64, LL);
    pack_w<<<cdiv(64 * 64, 256), 256, 0, stream>>>(oW1, oW1p, 64, 64, 1);

    // init state
    hipMemsetAsync(pos, 0, (size_t)BB * NN * 3 * 4, stream);
    hipMemsetAsync(cnts, 0, (size_t)NN * 4, stream);
    counts_kernel<<<cdiv(EE, 256), 256, 0, stream>>>(ei, cnts);
    g_kernel<<<cdiv(BB * HH, 256), 256, 0, stream>>>(z, Wg, bg, gbuf);
    hinit_kernel<<<cdiv(BB * NN * HH, 256), 256, 0, stream>>>(nf, Wn, bn, gbuf, hf, hb);

    for (int l = 0; l < LL; ++l) {
        hipMemsetAsync(agg, 0, (size_t)BB * NN * HH * 4, stream);
        hipMemsetAsync(pacc, 0, (size_t)BB * NN * 3 * 4, stream);
        edge_kernel<<<2048, 256, 0, stream>>>(
            hb, pos, ei,
            eW1p + (size_t)l * 64 * K1, eb1 + l * 64,
            eW2p + (size_t)l * 64 * 64, eb2 + l * 64,
            cW1p + (size_t)l * 64 * 64, cb1 + l * 64, cW2 + l * 64,
            agg, pacc);
        posupd_kernel<<<cdiv(BB * NN * 3, 256), 256, 0, stream>>>(pos, pacc, cnts);
        node_kernel<<<128, 256, 0, stream>>>(
            hb, agg, hf,
            nW1p + (size_t)l * 64 * 128, nb1 + l * 64,
            nW2p + (size_t)l * 64 * 64, nb2 + l * 64,
            lng + l * 64, lnb + l * 64);
    }

    out_kernel<<<128, 256, 0, stream>>>(hb, oW1p, ob1, oW2, ob2, out);
}
